// GNNModel_54949811585206
// MI455X (gfx1250) — matrix operand order repaired and checked by reading, not compiled, not run on hardware
//
#include <hip/hip_runtime.h>
#include <hip/hip_fp16.h>

typedef __attribute__((ext_vector_type(16))) _Float16 v16h;
typedef __attribute__((ext_vector_type(8)))  float    v8f;

// ---------------- degree counting ----------------
__global__ void k_degree(const int* __restrict__ e1, const int* __restrict__ e2,
                         float* __restrict__ cnt1, float* __restrict__ cnt2, int E) {
    int t = blockIdx.x * blockDim.x + threadIdx.x;
    if (t >= E) return;
    unsafeAtomicAdd(&cnt1[e1[t]], 1.0f);
    unsafeAtomicAdd(&cnt2[e2[t]], 1.0f);
}

__global__ void k_invert(float* __restrict__ v, int N) {
    int t = blockIdx.x * blockDim.x + threadIdx.x;
    if (t >= N) return;
    v[t] = 1.0f / fmaxf(v[t], 1.0f);
}

// ---------------- layer-1 (cin=1) ----------------
__global__ void k_edge_agg_scalar(const int* __restrict__ e1, const int* __restrict__ e2,
                                  const float* __restrict__ x1, const float* __restrict__ x2,
                                  float* __restrict__ agg1s, float* __restrict__ agg2s, int E) {
    int t = blockIdx.x * blockDim.x + threadIdx.x;
    if (t >= E) return;
    int a = e1[t], b = e2[t];
    unsafeAtomicAdd(&agg1s[a], x2[b]);   // messages x2 -> side 1
    unsafeAtomicAdd(&agg2s[b], x1[a]);   // messages x1 -> side 2
}

__global__ void k_node_l1(const float* __restrict__ aggs, const float* __restrict__ inv,
                          const float* __restrict__ x,
                          const float* __restrict__ Wl, const float* __restrict__ bl,
                          const float* __restrict__ Wr,
                          float* __restrict__ out, int N) {
    int t = blockIdx.x * blockDim.x + threadIdx.x;
    if (t >= N * 16) return;
    int i = t >> 4, j = t & 15;
    float m = aggs[i] * inv[i];
    float v = m * Wl[j] + bl[j] + x[i] * Wr[j];
    out[t] = fmaxf(v, 0.0f);
}

// ---------------- H=16 edge aggregation (scatter-add of 16-vectors) --------
// 4 threads per edge, float4 (b128) feature loads, 4 f32 atomics per thread.
__global__ void k_edge_agg_vec(const float* __restrict__ hsrc,
                               const int* __restrict__ srcIdx, const int* __restrict__ dstIdx,
                               float* __restrict__ agg, int E) {
    long long t = (long long)blockIdx.x * blockDim.x + threadIdx.x;
    if (t >= (long long)E * 4) return;
    int e = (int)(t >> 2);
    int q = (int)(t & 3) * 4;            // column base within the 16-vector
    int s = srcIdx[e];
    int d = dstIdx[e];
    const float4 m = *(const float4*)(hsrc + (long long)s * 16 + q);
    float* dst = agg + (long long)d * 16 + q;
    unsafeAtomicAdd(dst + 0, m.x);
    unsafeAtomicAdd(dst + 1, m.y);
    unsafeAtomicAdd(dst + 2, m.z);
    unsafeAtomicAdd(dst + 3, m.w);
}

// ---------------- fused SAGE node update via one 16x16x32 f16 WMMA ---------
// D = [agg*inv | hdst](16x32) x [Wl ; Wr](32x16), then +bias, ReLU.
__global__ void __launch_bounds__(256)
k_node_wmma(const float* __restrict__ agg,  const float* __restrict__ inv,
            const float* __restrict__ hdst,
            const float* __restrict__ Wl,   const float* __restrict__ bl,
            const float* __restrict__ Wr,
            float* __restrict__ out, int N) {
    const int lane = threadIdx.x & 31;
    const int wave = threadIdx.x >> 5;
    const int numTiles = (N + 15) >> 4;
    const int tile = blockIdx.x * 8 + wave;
    if (tile >= numTiles) return;          // wave-uniform: EXEC stays all-1s for WMMA

    const int  n      = lane & 15;
    const bool hiHalf = (lane >= 16);

    // B (32x16) 16-bit layout: lanes 0-15 K{0..7,16..23}; lanes 16-31 K{8..15,24..31}
    const int kb = hiHalf ? 8 : 0;         // rows 0..15 = Wl, rows 16..31 = Wr
    v16h b;
    #pragma unroll
    for (int j = 0; j < 8; ++j) {
        b[j]     = (_Float16)Wl[(kb + j) * 16 + n];
        b[8 + j] = (_Float16)Wr[(kb + j) * 16 + n];
    }

    // A (16x32) 16-bit layout: lanes 0-15 row=lane  K{0..7,16..23};
    //                          lanes 16-31 row=lane-16 K{8..15,24..31}
    int r = tile * 16 + n;
    if (r >= N) r = N - 1;                 // clamp reads; stores guarded below
    const int   c0 = hiHalf ? 8 : 0;
    const float s  = inv[r];
    v16h a;
    #pragma unroll
    for (int k = 0; k < 8; ++k) {
        a[k]     = (_Float16)(agg [r * 16 + c0 + k] * s);   // K 0..15  -> mean-agg path
        a[8 + k] = (_Float16)(hdst[r * 16 + c0 + k]);       // K 16..31 -> root path
    }

    v8f c = {};
    c = __builtin_amdgcn_wmma_f32_16x16x32_f16(false, a, false, b, (short)0, c, false, false);

    // C/D layout: vgpr v -> row v (lanes 0-15) / row v+8 (lanes 16-31), col = lane&15
    const float bias  = bl[n];
    const int   mbase = hiHalf ? 8 : 0;
    const bool  fullTile = (tile * 16 + 16 <= N);   // wave-uniform hot path
    if (fullTile) {
        #pragma unroll
        for (int v = 0; v < 8; ++v)
            out[(tile * 16 + mbase + v) * 16 + n] = fmaxf(c[v] + bias, 0.0f);
    } else {
        #pragma unroll
        for (int v = 0; v < 8; ++v) {
            int row = tile * 16 + mbase + v;
            if (row < N) out[row * 16 + n] = fmaxf(c[v] + bias, 0.0f);
        }
    }
}

// ---------------- final fc + sigmoid ----------------
__global__ void k_fc_sigmoid(const float* __restrict__ h, const float* __restrict__ W,
                             const float* __restrict__ bptr, float* __restrict__ out, int N) {
    int i = blockIdx.x * blockDim.x + threadIdx.x;
    if (i >= N) return;
    float acc = bptr[0];
    #pragma unroll
    for (int j = 0; j < 16; ++j) acc += h[i * 16 + j] * W[j];
    out[i] = 1.0f / (1.0f + __expf(-acc));
}

extern "C" void kernel_launch(void* const* d_in, const int* in_sizes, int n_in,
                              void* d_out, int out_size, void* d_ws, size_t ws_size,
                              hipStream_t stream) {
    const float* x1   = (const float*)d_in[0];
    const float* x2   = (const float*)d_in[1];
    const int*   eidx = (const int*)d_in[2];
    const int N1 = in_sizes[0];
    const int N2 = in_sizes[1];
    const int E  = in_sizes[2] / 2;
    const int* e1 = eidx;       // indexes x1
    const int* e2 = eidx + E;   // indexes x2

    const float* c1w2s_Wl = (const float*)d_in[3];
    const float* c1w2s_bl = (const float*)d_in[4];
    const float* c1w2s_Wr = (const float*)d_in[5];
    const float* c1s2w_Wl = (const float*)d_in[6];
    const float* c1s2w_bl = (const float*)d_in[7];
    const float* c1s2w_Wr = (const float*)d_in[8];
    const float* c2w2s_Wl = (const float*)d_in[9];
    const float* c2w2s_bl = (const float*)d_in[10];
    const float* c2w2s_Wr = (const float*)d_in[11];
    const float* c2s2w_Wl = (const float*)d_in[12];
    const float* c2s2w_bl = (const float*)d_in[13];
    const float* c2s2w_Wr = (const float*)d_in[14];
    const float* c3w2s_Wl = (const float*)d_in[15];
    const float* c3w2s_bl = (const float*)d_in[16];
    const float* c3w2s_Wr = (const float*)d_in[17];
    // d_in[18..20] = c3s2w (unused by reference output)
    const float* fc_W = (const float*)d_in[21];
    const float* fc_b = (const float*)d_in[22];

    // workspace layout (floats), ~78.4 MB total — fits in MI455X 192MB L2
    float* ws   = (float*)d_ws;
    float* inv1 = ws;                          // N1
    float* inv2 = inv1 + N1;                   // N2
    float* agg1 = inv2 + N2;                   // N1*16
    float* agg2 = agg1 + (size_t)N1 * 16;      // N2*16
    float* h1   = agg2 + (size_t)N2 * 16;      // N1*16
    float* h2   = h1   + (size_t)N1 * 16;      // N2*16
    float* g1   = h2   + (size_t)N2 * 16;      // N1*16
    float* g2   = g1   + (size_t)N1 * 16;      // N2*16

    const int TB = 256;
    const int gE   = (E + TB - 1) / TB;
    const long long E4 = (long long)E * 4;
    const int gE4  = (int)((E4 + TB - 1) / TB);
    const int tiles1 = (N1 + 15) / 16, tiles2 = (N2 + 15) / 16;
    const int gW1 = (tiles1 + 7) / 8,  gW2 = (tiles2 + 7) / 8;

    // degrees -> inverse counts
    hipMemsetAsync(inv1, 0, (size_t)(N1 + N2) * sizeof(float), stream);
    k_degree<<<gE, TB, 0, stream>>>(e1, e2, inv1, inv2, E);
    k_invert<<<((N1 + N2) + TB - 1) / TB, TB, 0, stream>>>(inv1, N1 + N2);

    // ---- layer 1 (cin=1, scalar messages) ----
    hipMemsetAsync(agg1, 0, (size_t)(N1 + N2) * 16 * sizeof(float), stream);
    k_edge_agg_scalar<<<gE, TB, 0, stream>>>(e1, e2, x1, x2, agg1, agg2, E);
    k_node_l1<<<(N1 * 16 + TB - 1) / TB, TB, 0, stream>>>(agg1, inv1, x1, c1w2s_Wl, c1w2s_bl, c1w2s_Wr, h1, N1);
    k_node_l1<<<(N2 * 16 + TB - 1) / TB, TB, 0, stream>>>(agg2, inv2, x2, c1s2w_Wl, c1s2w_bl, c1s2w_Wr, h2, N2);

    // ---- layer 2 (H=16, both directions) ----
    hipMemsetAsync(agg1, 0, (size_t)(N1 + N2) * 16 * sizeof(float), stream);
    k_edge_agg_vec<<<gE4, TB, 0, stream>>>(h2, e2, e1, agg1, E);
    k_edge_agg_vec<<<gE4, TB, 0, stream>>>(h1, e1, e2, agg2, E);
    k_node_wmma<<<gW1, TB, 0, stream>>>(agg1, inv1, h1, c2w2s_Wl, c2w2s_bl, c2w2s_Wr, g1, N1);
    k_node_wmma<<<gW2, TB, 0, stream>>>(agg2, inv2, h2, c2s2w_Wl, c2s2w_bl, c2s2w_Wr, g2, N2);

    // ---- layer 3 (only w2s direction feeds the output) ----
    hipMemsetAsync(agg1, 0, (size_t)N1 * 16 * sizeof(float), stream);
    k_edge_agg_vec<<<gE4, TB, 0, stream>>>(g2, e2, e1, agg1, E);
    k_node_wmma<<<gW1, TB, 0, stream>>>(agg1, inv1, g1, c3w2s_Wl, c3w2s_bl, c3w2s_Wr, h1, N1);

    // ---- fc + sigmoid ----
    k_fc_sigmoid<<<(N1 + TB - 1) / TB, TB, 0, stream>>>(h1, fc_W, fc_b, (float*)d_out, N1);
}
